// SetAbstraction_31061203485290
// MI455X (gfx1250) — matrix operand order repaired and checked by reading, not compiled, not run on hardware
//
#include <hip/hip_runtime.h>

// Problem constants (match reference)
#define NB   4
#define NPT  8192
#define CIN  64
#define COUT 128
#define KNB  32
#define MPT  2048
#define RAD2 0.01f   // 0.1^2

typedef __attribute__((ext_vector_type(16))) _Float16 v16h;
typedef __attribute__((ext_vector_type(8)))  float    v8f;
typedef int v4i __attribute__((vector_size(16)));

#if __has_builtin(__builtin_amdgcn_global_load_async_to_lds_b128) && \
    __has_builtin(__builtin_amdgcn_s_wait_asynccnt)
#define HAVE_ASYNC_LDS 1
#endif

// ---------------- WMMA tile loaders (16x16x32 f16, contiguous k = 16h + s for A and B) --
// A matrix 16x32 (MxK): lanes 0-15 rows 0-15 hold K 0..15; lanes 16-31
// hold K 16..31.
__device__ inline v16h wmma_load_A(const float* W, int ld, int row0, int k0, int lane) {
  int row = row0 + (lane & 15);
  int kh  = (lane >> 4) << 3;
  v16h a;
#pragma unroll
  for (int j = 0; j < 16; ++j) {
    int k = k0 + 2 * kh + j;
    a[j] = (_Float16)W[row * ld + k];
  }
  return a;
}

// B matrix 32x16 (KxN) from an LDS f16 tile stored [k*ld + n]:
// lanes 0-15 (N=lane) hold K 0..15, lanes 16-31 hold K 16..31.
__device__ inline v16h wmma_load_B(const _Float16* t, int ld, int k0, int n0, int lane) {
  int n  = n0 + (lane & 15);
  int kb = k0 + ((lane >> 4) << 4);
  v16h b;
#pragma unroll
  for (int j = 0; j < 16; ++j) b[j] = t[(kb + j) * ld + n];
  return b;
}

// Same, but from an f32 LDS tile (convert at load; async copies stage raw f32).
__device__ inline v16h wmma_load_B_f32(const float* t, int ld, int k0, int n0, int lane) {
  int n  = n0 + (lane & 15);
  int kb = k0 + ((lane >> 4) << 4);
  v16h b;
#pragma unroll
  for (int j = 0; j < 16; ++j) b[j] = (_Float16)t[(kb + j) * ld + n];
  return b;
}

__device__ inline v8f wmma_f16(v16h a, v16h b, v8f c) {
  return __builtin_amdgcn_wmma_f32_16x16x32_f16(false, a, false, b, (short)0, c, false, false);
}

// ---------------- Kernel 1: furthest point sampling (1 WG per batch) ---------
// Whole per-batch point cloud (96 KB) + mind (32 KB) resident in LDS: the 2047
// serial argmax steps become DS-latency bound instead of L2-latency bound.
// 128 KB dynamic LDS <= 320 KB/WGP on CDNA5.
__global__ __launch_bounds__(1024) void fps_kernel(const float* __restrict__ p,
                                                   int* __restrict__ idx) {
  extern __shared__ float fsm[];          // px | py | pz | mind  (4 * NPT floats)
  float* px   = fsm;
  float* py   = fsm + NPT;
  float* pz   = fsm + 2 * NPT;
  float* mind = fsm + 3 * NPT;
  __shared__ float wval[32];
  __shared__ int   widx[32];
  __shared__ int   curS;
  const int b = blockIdx.x;
  const int t = threadIdx.x;
  for (int n = t; n < NPT; n += 1024) {
    px[n] = p[(b * NPT + n) * 3 + 0];
    py[n] = p[(b * NPT + n) * 3 + 1];
    pz[n] = p[(b * NPT + n) * 3 + 2];
    mind[n] = 1e10f;
  }
  if (t == 0) { idx[b * MPT] = 0; curS = 0; }
  __syncthreads();
  int cur = 0;
  for (int s = 1; s < MPT; ++s) {
    const float lx = px[cur], ly = py[cur], lz = pz[cur];
    float bv = -1.0f; int bi = 0;
    for (int n = t; n < NPT; n += 1024) {
      float dx = px[n] - lx, dy = py[n] - ly, dz = pz[n] - lz;
      float d2 = dx * dx + dy * dy + dz * dz;
      float mv = fminf(mind[n], d2);
      mind[n] = mv;
      if (mv > bv) { bv = mv; bi = n; }       // ascending n: first-max tie-break
    }
    // wave32 argmax reduction (prefer smaller index on ties)
    for (int o = 16; o > 0; o >>= 1) {
      float ov = __shfl_down(bv, o, 32);
      int   oi = __shfl_down(bi, o, 32);
      if (ov > bv || (ov == bv && oi < bi)) { bv = ov; bi = oi; }
    }
    if ((t & 31) == 0) { wval[t >> 5] = bv; widx[t >> 5] = bi; }
    __syncthreads();
    if (t == 0) {
      float best = wval[0]; int bb = widx[0];
      for (int w = 1; w < 32; ++w)
        if (wval[w] > best || (wval[w] == best && widx[w] < bb)) { best = wval[w]; bb = widx[w]; }
      curS = bb; idx[b * MPT + s] = bb;
    }
    __syncthreads();
    cur = curS;
  }
}

// ---------------- Kernel 2: gather sampled positions -------------------------
__global__ void gather_newp_kernel(const float* __restrict__ p, const int* __restrict__ idx,
                                   float* __restrict__ newp) {
  int q = blockIdx.x * blockDim.x + threadIdx.x;   // 0 .. NB*MPT-1
  if (q >= NB * MPT) return;
  int b = q / MPT;
  int n = idx[q];
  newp[q * 3 + 0] = p[(b * NPT + n) * 3 + 0];
  newp[q * 3 + 1] = p[(b * NPT + n) * 3 + 1];
  newp[q * 3 + 2] = p[(b * NPT + n) * 3 + 2];
}

// ---------------- Kernel 3: identity = Ws @ f[:, idx] + bs  (WMMA) -----------
__global__ __launch_bounds__(256) void identity_kernel(const float* __restrict__ f,
                                                       const int* __restrict__ idx,
                                                       const float* __restrict__ Ws,
                                                       const float* __restrict__ bs,
                                                       float* __restrict__ ident) {
  __shared__ _Float16 fiT[CIN * 16];
  __shared__ int mI[16];
  const int b = blockIdx.y, m0 = blockIdx.x * 16;
  const int t = threadIdx.x, lane = t & 31, wave = t >> 5;
  if (t < 16) mI[t] = idx[b * MPT + m0 + t];
  __syncthreads();
#pragma unroll
  for (int i = 0; i < 4; ++i) {
    int e = t + 256 * i;                 // 1024 elements
    int c = e >> 4, n = e & 15;
    fiT[c * 16 + n] = (_Float16)f[(b * CIN + c) * NPT + mI[n]];
  }
  __syncthreads();
  v8f acc = {};
  acc = wmma_f16(wmma_load_A(Ws, CIN, 16 * wave, 0, lane),  wmma_load_B(fiT, 16, 0, 0, lane),  acc);
  acc = wmma_f16(wmma_load_A(Ws, CIN, 16 * wave, 32, lane), wmma_load_B(fiT, 16, 32, 0, lane), acc);
#pragma unroll
  for (int r = 0; r < 8; ++r) {
    int row = r + ((lane >> 4) << 3);
    int c = 16 * wave + row;
    ident[(b * COUT + c) * MPT + m0 + (lane & 15)] = acc[r] + bs[c];
  }
}

// ---------------- Kernel 4: fused pw1d (64->64 relu, 64->128)  (WMMA) --------
__global__ __launch_bounds__(256) void pw1d_kernel(const float* __restrict__ f,
                                                   const float* __restrict__ W10, const float* __restrict__ g10, const float* __restrict__ b10,
                                                   const float* __restrict__ W11, const float* __restrict__ g11, const float* __restrict__ b11,
                                                   float* __restrict__ h) {
  __shared__ __align__(16) float fT[CIN * 16];   // staged as f32 (async copy)
  __shared__ _Float16 h1T[CIN * 16];
  const int b = blockIdx.y, n0 = blockIdx.x * 16;
  const int t = threadIdx.x, lane = t & 31, wave = t >> 5;
#if defined(HAVE_ASYNC_LDS)
  {
    // 64 rows x 64 B: one b128 async global->LDS transfer per thread (ASYNCcnt).
    int c = t >> 2, ch = t & 3;
    const float* gsrc = f + (size_t)(b * CIN + c) * NPT + n0 + ch * 4;
    float*       ldst = &fT[c * 16 + ch * 4];
    __builtin_amdgcn_global_load_async_to_lds_b128((v4i*)gsrc, (v4i*)ldst, 0, 0);
    __builtin_amdgcn_s_wait_asynccnt(0);
  }
#else
#pragma unroll
  for (int i = 0; i < 4; ++i) {
    int e = t + 256 * i;
    int c = e >> 4, n = e & 15;
    fT[c * 16 + n] = f[(b * CIN + c) * NPT + n0 + n];
  }
#endif
  __syncthreads();
  if (wave < 4) {                        // layer 1: 64 output rows
    v8f acc = {};
    acc = wmma_f16(wmma_load_A(W10, CIN, 16 * wave, 0, lane),  wmma_load_B_f32(fT, 16, 0, 0, lane),  acc);
    acc = wmma_f16(wmma_load_A(W10, CIN, 16 * wave, 32, lane), wmma_load_B_f32(fT, 16, 32, 0, lane), acc);
#pragma unroll
    for (int r = 0; r < 8; ++r) {
      int row = 16 * wave + r + ((lane >> 4) << 3);
      float v = fmaxf(g10[row] * acc[r] + b10[row], 0.0f);
      h1T[row * 16 + (lane & 15)] = (_Float16)v;
    }
  }
  __syncthreads();
  {                                      // layer 2: 128 output rows, no relu
    v8f acc = {};
    acc = wmma_f16(wmma_load_A(W11, CIN, 16 * wave, 0, lane),  wmma_load_B(h1T, 16, 0, 0, lane),  acc);
    acc = wmma_f16(wmma_load_A(W11, CIN, 16 * wave, 32, lane), wmma_load_B(h1T, 16, 32, 0, lane), acc);
#pragma unroll
    for (int r = 0; r < 8; ++r) {
      int row = 16 * wave + r + ((lane >> 4) << 3);
      h[(b * COUT + row) * NPT + n0 + (lane & 15)] = g11[row] * acc[r] + b11[row];
    }
  }
}

// ---------------- Kernel 5: ball query (first-K in index order) --------------
__global__ __launch_bounds__(256) void ballq_kernel(const float* __restrict__ newp,
                                                    const float* __restrict__ p,
                                                    int* __restrict__ gidx) {
  const int lane = threadIdx.x & 31;
  const int q = blockIdx.x * 8 + (threadIdx.x >> 5);     // one wave per query
  const int b = q / MPT;
  const float qx = newp[q * 3 + 0], qy = newp[q * 3 + 1], qz = newp[q * 3 + 2];
  int count = 0, f0 = 0;
  for (int base = 0; base < NPT && count < KNB; base += 32) {
    if (base + 256 < NPT)
      __builtin_prefetch(&p[(size_t)(b * NPT + base + 256) * 3], 0, 0);  // global_prefetch_b8
    int n = base + lane;
    float dx = p[(b * NPT + n) * 3 + 0] - qx;
    float dy = p[(b * NPT + n) * 3 + 1] - qy;
    float dz = p[(b * NPT + n) * 3 + 2] - qz;
    bool in = (dx * dx + dy * dy + dz * dz) <= RAD2;
    unsigned mask = (unsigned)__ballot(in);
    if (count == 0 && mask) f0 = base + (__ffs((int)mask) - 1);
    int pos = count + __popc(mask & ((1u << lane) - 1u));
    if (in && pos < KNB) gidx[q * KNB + pos] = n;
    count += __popc(mask);
  }
  if (count < KNB)
    for (int j = count + lane; j < KNB; j += 32) gidx[q * KNB + j] = f0;
}

// ---------------- Kernel 6: fused pw2d chain + fj gather + K-max + skip ------
__global__ __launch_bounds__(256) void tail_kernel(const float* __restrict__ p,
                                                   const float* __restrict__ newp,
                                                   const int*   __restrict__ gidx,
                                                   const float* __restrict__ h,
                                                   const float* __restrict__ ident,
                                                   const float* __restrict__ W20, const float* __restrict__ g20, const float* __restrict__ b20,
                                                   const float* __restrict__ W21, const float* __restrict__ g21, const float* __restrict__ b21,
                                                   const float* __restrict__ W22, const float* __restrict__ g22, const float* __restrict__ b22,
                                                   float* __restrict__ out) {
  __shared__ float    dpL[3 * KNB];
  __shared__ _Float16 pe1T[32 * KNB];
  __shared__ _Float16 pe2T[32 * KNB];
  __shared__ int      gI[KNB];
  const int m = blockIdx.x, b = blockIdx.y;
  const int q = b * MPT + m;
  const int t = threadIdx.x, lane = t & 31, wave = t >> 5;
  if (t < KNB) {
    int n = gidx[q * KNB + t];
    gI[t] = n;
    dpL[0 * KNB + t] = p[(b * NPT + n) * 3 + 0] - newp[q * 3 + 0];
    dpL[1 * KNB + t] = p[(b * NPT + n) * 3 + 1] - newp[q * 3 + 1];
    dpL[2 * KNB + t] = p[(b * NPT + n) * 3 + 2] - newp[q * 3 + 2];
  }
  __syncthreads();
  // layer 1 (3->32): tiny K, scalar VALU
#pragma unroll
  for (int i = 0; i < 4; ++i) {
    int e = t + 256 * i;                 // 1024 entries
    int o = e >> 5, k = e & 31;
    float s = W20[o * 3 + 0] * dpL[0 * KNB + k] + W20[o * 3 + 1] * dpL[1 * KNB + k] +
              W20[o * 3 + 2] * dpL[2 * KNB + k];
    pe1T[o * KNB + k] = (_Float16)fmaxf(g20[o] * s + b20[o], 0.0f);
  }
  __syncthreads();
  // layer 2 (32->32) via WMMA: 2 row-tiles x 2 n-tiles, waves 0-3
  if (wave < 4) {
    int rt = wave >> 1, nt = wave & 1;
    v8f acc = {};
    acc = wmma_f16(wmma_load_A(W21, 32, 16 * rt, 0, lane), wmma_load_B(pe1T, KNB, 0, 16 * nt, lane), acc);
#pragma unroll
    for (int r = 0; r < 8; ++r) {
      int row = 16 * rt + r + ((lane >> 4) << 3);
      pe2T[row * KNB + 16 * nt + (lane & 15)] =
          (_Float16)fmaxf(g21[row] * acc[r] + b21[row], 0.0f);
    }
  }
  __syncthreads();
  // layer 3 (32->128) via WMMA + fj add + max over K + identity + relu
  v16h a = wmma_load_A(W22, 32, 16 * wave, 0, lane);
  v8f acc0 = {}, acc1 = {};
  acc0 = wmma_f16(a, wmma_load_B(pe2T, KNB, 0, 0, lane),  acc0);
  acc1 = wmma_f16(a, wmma_load_B(pe2T, KNB, 0, 16, lane), acc1);
  const int k0 = gI[lane & 15], k1 = gI[16 + (lane & 15)];
#pragma unroll
  for (int r = 0; r < 8; ++r) {
    int row = r + ((lane >> 4) << 3);
    int c = 16 * wave + row;
    float ga = g22[c], bb = b22[c];
    float v0 = fmaxf(ga * acc0[r] + bb, 0.0f) + h[(b * COUT + c) * NPT + k0];
    float v1 = fmaxf(ga * acc1[r] + bb, 0.0f) + h[(b * COUT + c) * NPT + k1];
    float v = fmaxf(v0, v1);
    for (int o = 8; o > 0; o >>= 1) v = fmaxf(v, __shfl_xor(v, o, 32));  // max over 16 cols
    if ((lane & 15) == 0)
      out[(b * COUT + c) * MPT + m] = fmaxf(v + ident[(b * COUT + c) * MPT + m], 0.0f);
  }
}

// ---------------- host-side launcher -----------------------------------------
extern "C" void kernel_launch(void* const* d_in, const int* in_sizes, int n_in,
                              void* d_out, int out_size, void* d_ws, size_t ws_size,
                              hipStream_t stream) {
  (void)in_sizes; (void)n_in; (void)out_size; (void)ws_size;
  const float* p   = (const float*)d_in[0];
  const float* f   = (const float*)d_in[1];
  const float* W10 = (const float*)d_in[2];
  const float* g10 = (const float*)d_in[3];
  const float* b10 = (const float*)d_in[4];
  const float* W11 = (const float*)d_in[5];
  const float* g11 = (const float*)d_in[6];
  const float* b11 = (const float*)d_in[7];
  const float* W20 = (const float*)d_in[8];
  const float* g20 = (const float*)d_in[9];
  const float* b20 = (const float*)d_in[10];
  const float* W21 = (const float*)d_in[11];
  const float* g21 = (const float*)d_in[12];
  const float* b21 = (const float*)d_in[13];
  const float* W22 = (const float*)d_in[14];
  const float* g22 = (const float*)d_in[15];
  const float* b22 = (const float*)d_in[16];
  const float* Ws  = (const float*)d_in[17];
  const float* bs  = (const float*)d_in[18];
  float* out = (float*)d_out;

  // workspace layout (~21.2 MB)
  char* base = (char*)d_ws;
  int*   idx   = (int*)base;                       base += (size_t)NB * MPT * sizeof(int);
  float* newp  = (float*)base;                     base += (size_t)NB * MPT * 3 * sizeof(float);
  int*   gidx  = (int*)base;                       base += (size_t)NB * MPT * KNB * sizeof(int);
  float* ident = (float*)base;                     base += (size_t)NB * COUT * MPT * sizeof(float);
  float* h     = (float*)base;                     base += (size_t)NB * COUT * NPT * sizeof(float);

  fps_kernel<<<NB, 1024, 4 * NPT * sizeof(float), stream>>>(p, idx);   // 128 KB LDS
  gather_newp_kernel<<<(NB * MPT + 255) / 256, 256, 0, stream>>>(p, idx, newp);
  identity_kernel<<<dim3(MPT / 16, NB), 256, 0, stream>>>(f, idx, Ws, bs, ident);
  pw1d_kernel<<<dim3(NPT / 16, NB), 256, 0, stream>>>(f, W10, g10, b10, W11, g11, b11, h);
  ballq_kernel<<<(NB * MPT) / 8, 256, 0, stream>>>(newp, p, gidx);
  tail_kernel<<<dim3(MPT, NB), 256, 0, stream>>>(p, newp, gidx, h, ident,
                                                 W20, g20, b20, W21, g21, b21,
                                                 W22, g22, b22, out);
}
